// LAHead15b_61108794687937
// MI455X (gfx1250) — matrix operand order repaired and checked by reading, not compiled, not run on hardware
//
#include <hip/hip_runtime.h>

// ---- problem constants -----------------------------------------------------
#define B_   2
#define H_   512
#define W_   512
#define HW_  (H_*W_)      // 262144
#define CC_  128
#define CH_  64
#define CO_  64
#define HC_  128
#define WC_  128
#define HWC_ (HC_*WC_)    // 16384

typedef __attribute__((ext_vector_type(16))) _Float16 v16h;
typedef __attribute__((ext_vector_type(8)))  float    v8f;
typedef __attribute__((ext_vector_type(4)))  float    v4f;

union V16H { v16h v; unsigned u[8]; };

// A-matrix (16x32 f16, M x K) fragment from row-major f16 (row stride ld).
// ISA 7.12.2: lane m=L&15, half=L>>4; VGPR j<4 -> K=half*8+2j, j>=4 -> K=16+half*8+2(j-4).
__device__ __forceinline__ v16h load_a_frag_g(const _Float16* a, int ld, int m0,
                                              int k0, int lane) {
  int m = m0 + (lane & 15);
  int half = lane >> 4;
  V16H r;
#pragma unroll
  for (int j = 0; j < 8; ++j) {
    int k = (j < 4) ? (half * 8 + 2 * j) : (16 + half * 8 + 2 * (j - 4));
    r.u[j] = *(const unsigned*)(a + m * ld + k0 + k);
  }
  return r.v;
}

// B-matrix (32x16 f16, K x N) fragment from LDS stored n-major: bs[n*ld + k]
// lane n=L&15, half=L>>4; VGPR j<4 -> K=half*8+2j, j>=4 -> K=16+half*8+2(j-4).
__device__ __forceinline__ v16h load_b_frag_lds(const _Float16* bs, int ld, int n0,
                                                int k0, int lane) {
  int n = n0 + (lane & 15);
  int half = lane >> 4;
  V16H r;
#pragma unroll
  for (int j = 0; j < 8; ++j) {
    int k = (j < 4) ? (half * 8 + 2 * j) : (16 + half * 8 + 2 * (j - 4));
    r.u[j] = *(const unsigned*)(bs + n * ld + k0 + k);
  }
  return r.v;
}

// ---- kernel 0: pack weights to f16 ----------------------------------------
__global__ void __launch_bounds__(256)
k_convert(const float* __restrict__ wq2, const float* __restrict__ wk2,
          const float* __restrict__ wv,
          _Float16* __restrict__ wq2h, _Float16* __restrict__ wk2h,
          _Float16* __restrict__ wvch) {
  int t = blockIdx.x * 256 + threadIdx.x;       // 16384 threads total
  if (t < 4096) {
    wq2h[t] = (_Float16)wq2[t];
  } else if (t < 8192) {
    int i = t - 4096;
    wk2h[i] = (_Float16)wk2[i];
  } else {
    int i = t - 8192;                            // 64*128 coarse slice of wv
    int o = i >> 7, c = i & 127;
    wvch[i] = (_Float16)wv[o * 131 + c];
  }
}

// ---- kernel 1: coarse-resolution projection yc = Wv[:, :128] @ coarse ------
__global__ void __launch_bounds__(128)
k_coarse_gemm(const float* __restrict__ coarse, const _Float16* __restrict__ wvch,
              float* __restrict__ yc) {
  __shared__ __attribute__((aligned(16))) _Float16 Bls[16][136];  // [n][c], padded
  int b  = blockIdx.x >> 10;
  int p0 = (blockIdx.x & 1023) << 4;
  int t  = threadIdx.x;

  int n = t & 15, cg = t >> 4;                  // cg in 0..7
#pragma unroll
  for (int i = 0; i < 16; ++i) {
    int c = cg * 16 + i;
    Bls[n][c] = (_Float16)coarse[(size_t)((b * CC_ + c) * HWC_) + p0 + n];
  }
  __syncthreads();

  int wid = t >> 5, lane = t & 31;
  v8f acc = {};
#pragma unroll
  for (int kc = 0; kc < 4; ++kc) {
    v16h af = load_a_frag_g(wvch, 128, wid * 16, kc * 32, lane);
    v16h bf = load_b_frag_lds(&Bls[0][0], 136, 0, kc * 32, lane);
    acc = __builtin_amdgcn_wmma_f32_16x16x32_f16(false, af, false, bf,
                                                 (short)0, acc, false, false);
  }
  int half = lane >> 4, nn = lane & 15;
#pragma unroll
  for (int r = 0; r < 8; ++r) {
    int m = wid * 16 + half * 8 + r;            // D layout: VGPR r -> M=half*8+r
    yc[(size_t)((b * CO_ + m) * HWC_) + p0 + nn] = acc[r];
  }
}

// ---- kernel 2: full-res Q/K (WMMA 64x64 layer) + V (bilinear + raw slice) --
__global__ void __launch_bounds__(256)
k_qkv(const float* __restrict__ raw,
      const float* __restrict__ wq1, const float* __restrict__ bq1,
      const float* __restrict__ bq2,
      const float* __restrict__ wk1, const float* __restrict__ bk1,
      const float* __restrict__ bk2,
      const float* __restrict__ wv,  const float* __restrict__ bv,
      const _Float16* __restrict__ wq2h, const _Float16* __restrict__ wk2h,
      const float* __restrict__ yc,
      float* __restrict__ xq, float* __restrict__ xk, float* __restrict__ xv) {
  __shared__ __attribute__((aligned(16))) _Float16 hq[64][68];   // [pixel][j]
  __shared__ __attribute__((aligned(16))) _Float16 hk[64][68];

  int blk = blockIdx.x;                          // B*H*(W/64) = 8192
  int b   = blk >> 12;
  int rem = blk & 4095;
  int y   = rem >> 3;
  int x0  = (rem & 7) << 6;
  int t   = threadIdx.x;

  // --- first MLP layer (3 -> 64), VALU; hidden vectors to LDS as f16 -------
  {
    int pix = t & 63, jg = t >> 6;               // 16 j's per thread
    int x = x0 + pix;
    size_t rbase = (size_t)(b * 3) * HW_ + (size_t)y * W_ + x;
    float r0 = raw[rbase], r1 = raw[rbase + HW_], r2 = raw[rbase + 2 * HW_];
#pragma unroll
    for (int i = 0; i < 16; ++i) {
      int j = jg * 16 + i;
      float hqv = fmaxf(bq1[j] + wq1[j*3]*r0 + wq1[j*3+1]*r1 + wq1[j*3+2]*r2, 0.f);
      float hkv = fmaxf(bk1[j] + wk1[j*3]*r0 + wk1[j*3+1]*r1 + wk1[j*3+2]*r2, 0.f);
      hq[pix][j] = (_Float16)hqv;
      hk[pix][j] = (_Float16)hkv;
    }
  }
  __syncthreads();

  // --- second MLP layer (64 -> 64) via WMMA: 8 waves = 4 M-tiles x {q,k} ----
  {
    int wid = t >> 5, lane = t & 31;
    int sel = wid >> 2, mt = wid & 3;
    const _Float16* w2 = sel ? wk2h : wq2h;
    const float*    b2 = sel ? bk2  : bq2;
    const _Float16* hs = sel ? &hk[0][0] : &hq[0][0];
    float* dst = sel ? xk : xq;

    v16h a0 = load_a_frag_g(w2, 64, mt * 16, 0, lane);
    v16h a1 = load_a_frag_g(w2, 64, mt * 16, 32, lane);

    int half = lane >> 4, nn = lane & 15;
    float bias[8];
#pragma unroll
    for (int r = 0; r < 8; ++r) bias[r] = b2[mt * 16 + half * 8 + r];

#pragma unroll
    for (int nt = 0; nt < 4; ++nt) {
      v16h b0 = load_b_frag_lds(hs, 68, nt * 16, 0, lane);
      v16h b1 = load_b_frag_lds(hs, 68, nt * 16, 32, lane);
      v8f acc = {};
      acc = __builtin_amdgcn_wmma_f32_16x16x32_f16(false, a0, false, b0,
                                                   (short)0, acc, false, false);
      acc = __builtin_amdgcn_wmma_f32_16x16x32_f16(false, a1, false, b1,
                                                   (short)0, acc, false, false);
      int x = x0 + nt * 16 + nn;
      size_t obase = ((size_t)(b * H_ + y) * W_ + x) * 64 + mt * 16 + half * 8;
      v4f lo = { acc[0]+bias[0], acc[1]+bias[1], acc[2]+bias[2], acc[3]+bias[3] };
      v4f hi = { acc[4]+bias[4], acc[5]+bias[5], acc[6]+bias[6], acc[7]+bias[7] };
      *(v4f*)(dst + obase)     = lo;
      *(v4f*)(dst + obase + 4) = hi;
    }
  }

  // --- V: bilinear(yc) + wv[:,128:131] @ raw + bv --------------------------
  {
    int ch = t & 63, pg = t >> 6;
    float fy = (y + 0.5f) * 0.25f - 0.5f;        // half-pixel-center bilinear
    int   iy0 = (int)floorf(fy);
    float wy  = fy - (float)iy0;
    int y0c = min(max(iy0, 0), HC_ - 1), y1c = min(max(iy0 + 1, 0), HC_ - 1);
    float wvr0 = wv[ch*131 + 128], wvr1 = wv[ch*131 + 129], wvr2 = wv[ch*131 + 130];
    float bvv  = bv[ch];
    const float* ycb = yc + (size_t)(b * CO_ + ch) * HWC_;
#pragma unroll
    for (int i = 0; i < 16; ++i) {
      int pix = pg * 16 + i;
      int x = x0 + pix;
      float fx = (x + 0.5f) * 0.25f - 0.5f;
      int   ix0 = (int)floorf(fx);
      float wx  = fx - (float)ix0;
      int x0c = min(max(ix0, 0), WC_ - 1), x1c = min(max(ix0 + 1, 0), WC_ - 1);
      float v00 = ycb[y0c * WC_ + x0c], v01 = ycb[y0c * WC_ + x1c];
      float v10 = ycb[y1c * WC_ + x0c], v11 = ycb[y1c * WC_ + x1c];
      float up = (1.f - wy) * ((1.f - wx) * v00 + wx * v01)
               +        wy  * ((1.f - wx) * v10 + wx * v11);
      size_t rbase = (size_t)(b * 3) * HW_ + (size_t)y * W_ + x;
      float r0 = raw[rbase], r1 = raw[rbase + HW_], r2 = raw[rbase + 2 * HW_];
      xv[((size_t)(b * H_ + y) * W_ + x) * 64 + ch] =
          up + bvv + wvr0 * r0 + wvr1 * r1 + wvr2 * r2;
    }
  }
}

// ---- kernel 3: 3x3 dilated (dil=2) local attention -------------------------
// One block = 64 pixels of one row, 64 threads (1 thread/pixel).
// K halo tile (3 rows x 68 cols x 64 ch, f32 = 52,224 B) is staged into LDS
// with CDNA5 async global->LDS loads (ASYNCcnt), then scores read via DS.
__global__ void __launch_bounds__(64)
k_attn(const float* __restrict__ xq, const float* __restrict__ xk,
       const float* __restrict__ xv, float* __restrict__ out) {
  __shared__ __attribute__((aligned(16))) float kt[3][68][64];

  int t   = threadIdx.x;
  int pid = blockIdx.x * 64 + t;                 // b*HW + y*W + x
  int b   = pid >> 18;
  int rem = pid & (HW_ - 1);
  int y = rem >> 9, x = rem & 511;
  int x0 = x & ~63;                              // block-aligned row segment
  int lx = x & 63;

  // ---- async stage of K tile: rows {y-2,y,y+2}, cols x0-2 .. x0+65 --------
  unsigned ldsbase = (unsigned)(unsigned long long)(void*)&kt[0][0][0];
  // 3 rows * 68 px * 16 b128-chunks = 3264 chunks; 51 per thread.
#pragma unroll 1
  for (int ci = t; ci < 3 * 68 * 16; ci += 64) {
    int r  = ci / (68 * 16);
    int rm = ci % (68 * 16);
    int px = rm >> 4;
    int c4 = (rm & 15) << 2;                     // channel (x4 floats = 16B)
    int yy = y + r * 2 - 2;
    int xx = x0 - 2 + px;
    if (((unsigned)yy < (unsigned)H_) && ((unsigned)xx < (unsigned)W_)) {
      const float* gp = xk + ((size_t)(b * H_ + yy) * W_ + xx) * 64 + c4;
      unsigned lo = ldsbase + (unsigned)(((r * 68 + px) * 64 + c4) * 4);
      asm volatile("global_load_async_to_lds_b128 %0, %1, off"
                   :: "v"(lo), "v"(gp) : "memory");
    }
  }
  asm volatile("s_wait_asynccnt 0x0" ::: "memory");
  __syncthreads();

  const float* qp = xq + (size_t)pid * 64;
  const float* vp[9];
  bool  ok[9];
  float s[9];
#pragma unroll
  for (int n = 0; n < 9; ++n) {
    int dy = (n / 3) * 2 - 2, dx = (n % 3) * 2 - 2;
    int yy = y + dy, xx = x + dx;
    bool inb = ((unsigned)yy < (unsigned)H_) && ((unsigned)xx < (unsigned)W_);
    ok[n] = inb;
    size_t nb = ((size_t)(b * H_ + (inb ? yy : y)) * W_ + (inb ? xx : x)) * 64;
    vp[n] = xv + nb;
    s[n] = 0.f;                                  // OOB score stays 0 (zero-pad K)
  }

  // pass 1: scores, K from LDS tile
#pragma unroll 4
  for (int c = 0; c < 64; c += 4) {
    v4f q4 = *(const v4f*)(qp + c);
#pragma unroll
    for (int n = 0; n < 9; ++n) {
      if (ok[n]) {
        int r  = n / 3;
        int dx = (n % 3) * 2 - 2;
        v4f k4 = *(const v4f*)&kt[r][lx + 2 + dx][c];
        s[n] += q4[0]*k4[0] + q4[1]*k4[1] + q4[2]*k4[2] + q4[3]*k4[3];
      }
    }
  }

  // softmax over all 9 (OOB included with score 0, as in reference)
  float mx = s[0];
#pragma unroll
  for (int n = 1; n < 9; ++n) mx = fmaxf(mx, s[n]);
  float wgt[9], wsum = 0.f;
#pragma unroll
  for (int n = 0; n < 9; ++n) { wgt[n] = __expf(s[n] - mx); wsum += wgt[n]; }
  float inv = 1.f / wsum;
#pragma unroll
  for (int n = 0; n < 9; ++n) wgt[n] *= inv;

  // pass 2: weighted V (global, single use), NCHW output
  size_t ob = (size_t)(b * CO_) * HW_ + (size_t)y * W_ + x;
#pragma unroll 2
  for (int c = 0; c < 64; c += 4) {
    v4f acc = {};
#pragma unroll
    for (int n = 0; n < 9; ++n) {
      if (ok[n]) {
        v4f v4 = *(const v4f*)(vp[n] + c);
        acc += wgt[n] * v4;
      }
    }
    out[ob + (size_t)(c + 0) * HW_] = acc[0];
    out[ob + (size_t)(c + 1) * HW_] = acc[1];
    out[ob + (size_t)(c + 2) * HW_] = acc[2];
    out[ob + (size_t)(c + 3) * HW_] = acc[3];
  }
}

// ---- host launch -----------------------------------------------------------
extern "C" void kernel_launch(void* const* d_in, const int* in_sizes, int n_in,
                              void* d_out, int out_size, void* d_ws, size_t ws_size,
                              hipStream_t stream) {
  const float* coarse = (const float*)d_in[0];
  const float* raw    = (const float*)d_in[1];
  const float* wq1 = (const float*)d_in[2];
  const float* bq1 = (const float*)d_in[3];
  const float* wq2 = (const float*)d_in[4];
  const float* bq2 = (const float*)d_in[5];
  const float* wk1 = (const float*)d_in[6];
  const float* bk1 = (const float*)d_in[7];
  const float* wk2 = (const float*)d_in[8];
  const float* bk2 = (const float*)d_in[9];
  const float* wv  = (const float*)d_in[10];
  const float* bv  = (const float*)d_in[11];
  float* out = (float*)d_out;

  char* ws = (char*)d_ws;
  _Float16* wq2h = (_Float16*)(ws);                       //  8 KB
  _Float16* wk2h = (_Float16*)(ws + 8192);                //  8 KB
  _Float16* wvch = (_Float16*)(ws + 16384);               // 16 KB
  float* yc = (float*)(ws + ((size_t)1  << 20));          //  8 MiB
  float* xq = (float*)(ws + ((size_t)16 << 20));          // 128 MiB, NHWC
  float* xk = (float*)(ws + ((size_t)144 << 20));         // 128 MiB, NHWC
  float* xv = (float*)(ws + ((size_t)272 << 20));         // 128 MiB, NHWC

  k_convert<<<64, 256, 0, stream>>>(wq2, wk2, wv, wq2h, wk2h, wvch);
  k_coarse_gemm<<<B_ * (HWC_ / 16), 128, 0, stream>>>(coarse, wvch, yc);
  k_qkv<<<B_ * H_ * (W_ / 64), 256, 0, stream>>>(raw, wq1, bq1, bq2, wk1, bk1,
                                                 bk2, wv, bv, wq2h, wk2h, yc,
                                                 xq, xk, xv);
  k_attn<<<(B_ * HW_) / 64, 64, 0, stream>>>(xq, xk, xv, out);
}
